// CustomAttention_65231963291955
// MI455X (gfx1250) — compile-verified, matrix operand order checked by reading, not run on hardware
//
#include <hip/hip_runtime.h>

// ---------------------------------------------------------------------------
// CDNA5 (gfx1250) fused attention block, bf16 WMMA everywhere.
// B=4, S=2048, D=768, H=12, HD=64.
//  - weights pre-transposed once at bf16 conversion (no LDS transposes)
//  - V produced in [B,H,HD,S] layout by the QKV epilogue
//  - GLOBAL_LOAD_ASYNC_TO_LDS_B128: signature probed = 4 args,
//    (int4* src[global], int4* dst[lds], imm offset, imm cpol)
// ---------------------------------------------------------------------------

typedef __bf16 bf16_t;
typedef bf16_t v16bf __attribute__((ext_vector_type(16)));
typedef float  v8f   __attribute__((ext_vector_type(8)));
typedef int    v4i   __attribute__((vector_size(16)));

constexpr int B_  = 4;
constexpr int S_  = 2048;
constexpr int D_  = 768;
constexpr int H_  = 12;
constexpr int HD_ = 64;
constexpr int BS  = B_ * S_;      // 8192
constexpr int N3D = 3 * D_;       // 2304

#define AS1 __attribute__((address_space(1)))
#define AS3 __attribute__((address_space(3)))

// integer round-trips avoid illegal direct addrspace casts
#define GA(p) ((AS1 v4i*)(__UINTPTR_TYPE__)(p))
#define LA(p) ((AS3 v4i*)(unsigned int)(__UINTPTR_TYPE__)(p))

template <int N> struct prio : prio<N - 1> {};
template <>      struct prio<0> {};

#if __has_builtin(__builtin_amdgcn_global_load_async_to_lds_b128)
// p3: (AS1 src, AS3 dst, imm, imm)   <- known-good per round-4 probe
template <typename S, typename D>
__device__ __forceinline__ auto async_b128(S s, D d, prio<3>)
    -> decltype(__builtin_amdgcn_global_load_async_to_lds_b128(GA(s), LA(d), 0, 0), void()) {
  __builtin_amdgcn_global_load_async_to_lds_b128(GA(s), LA(d), 0, 0);
}
// p2: (AS1 src, generic dst, imm, imm)
template <typename S, typename D>
__device__ __forceinline__ auto async_b128(S s, D d, prio<2>)
    -> decltype(__builtin_amdgcn_global_load_async_to_lds_b128(GA(s), d, 0, 0), void()) {
  __builtin_amdgcn_global_load_async_to_lds_b128(GA(s), d, 0, 0);
}
// p1: (generic src, generic dst, imm, imm)
template <typename S, typename D>
__device__ __forceinline__ auto async_b128(S s, D d, prio<1>)
    -> decltype(__builtin_amdgcn_global_load_async_to_lds_b128(s, d, 0, 0), void()) {
  __builtin_amdgcn_global_load_async_to_lds_b128(s, d, 0, 0);
}
#endif
// p0: synchronous 16B copy fallback
template <typename S, typename D>
__device__ __forceinline__ void async_b128(S s, D d, prio<0>) {
  *(ulonglong2*)d = *(const ulonglong2*)s;
}

__device__ __forceinline__ void copy16_async(const bf16_t* src, bf16_t* dst) {
  async_b128((v4i*)src, (v4i*)dst, prio<3>{});
}

__device__ __forceinline__ void wait_async0() {
#if __has_builtin(__builtin_amdgcn_s_wait_asynccnt)
  __builtin_amdgcn_s_wait_asynccnt(0);
#else
  asm volatile("s_wait_asynccnt 0x0" ::: "memory");
#endif
}

// ---------------------------------------------------------------------------
// A-layout fragment load (ISA 7.12.2, 16-bit A matrix 16x32):
//   lane l holds row m = l&15; kbase = (l>>4)*8
//   elements 0..7  -> K = kbase+0 .. kbase+7
//   elements 8..15 -> K = 16+kbase .. 16+kbase+7
// B fragments (32x16, K x N) use the identical pattern when the tile is
// stored transposed (N x K row-major).
// ---------------------------------------------------------------------------
__device__ __forceinline__ v16bf frag_a(const bf16_t* __restrict__ t, int ld) {
  const int l  = threadIdx.x & 31;
  const int m  = l & 15;
  const int kb = (l >> 4) << 3;
  const bf16_t* r = t + m * ld;
  v16bf f;
#pragma unroll
  for (int h = 0; h < 8; ++h) f[h] = r[kb + h];
#pragma unroll
  for (int h = 0; h < 8; ++h) f[8 + h] = r[16 + kb + h];
  return f;
}

__device__ __forceinline__ v8f wmma_bf16(v16bf a, v16bf b, v8f c) {
  return __builtin_amdgcn_wmma_f32_16x16x32_bf16(false, a, false, b,
                                                 (short)0, c, false, false);
}

// ---------------------------------------------------------------------------
// fp32 -> bf16 conversion (x4 per thread)
// ---------------------------------------------------------------------------
__global__ void cvt_bf16_kernel(const float* __restrict__ x,
                                bf16_t* __restrict__ y, int n) {
  int i = (blockIdx.x * blockDim.x + threadIdx.x) * 4;
  if (i + 3 < n) {
    float4 v = *(const float4*)(x + i);
    y[i + 0] = (bf16_t)v.x;
    y[i + 1] = (bf16_t)v.y;
    y[i + 2] = (bf16_t)v.z;
    y[i + 3] = (bf16_t)v.w;
  }
}

// fp32 [K,N] -> bf16 transposed [N,K] (one-time; read coalesced)
__global__ void cvt_bf16_tr_kernel(const float* __restrict__ x,
                                   bf16_t* __restrict__ y, int K, int N) {
  int i = blockIdx.x * blockDim.x + threadIdx.x;
  if (i < K * N) {
    int k = i / N, n = i - k * N;
    y[(size_t)n * K + k] = (bf16_t)x[i];
  }
}

// ---------------------------------------------------------------------------
// Shared GEMM core: 128x128 output tile per 256-thread block (8 waves),
// K stepped by 32.  Both operands row-major-K strips:
//   A  [M,K]  rows m0..m0+128
//   Wt [N,K]  rows n0..n0+128   (weight pre-transposed in convert pass)
// Wave w owns rows [16w,16w+16) x 128 cols -> 8 f32 accumulators.
// Tiles staged into LDS via async-to-LDS when available.
// ---------------------------------------------------------------------------
template <int KD>
__device__ __forceinline__ void gemm_core(const bf16_t* __restrict__ A,
                                          const bf16_t* __restrict__ Wt,
                                          bf16_t* As, bf16_t* Bs,
                                          int m0, int n0, v8f acc[8]) {
  const int t = threadIdx.x;
  const int w = t >> 5;
  const int r = t >> 1;            // 0..127
  const int c = (t & 1) * 16;      // 0 or 16
  for (int ks = 0; ks < KD / 32; ++ks) {
    {  // A tile rows m0..+128, cols ks*32..+32 (16 bf16 / thread)
      const bf16_t* src = A + (size_t)(m0 + r) * KD + ks * 32 + c;
      bf16_t* dst = As + r * 32 + c;
      copy16_async(src, dst);
      copy16_async(src + 8, dst + 8);
    }
    {  // Wt tile rows n0..+128, cols ks*32..+32
      const bf16_t* src = Wt + (size_t)(n0 + r) * KD + ks * 32 + c;
      bf16_t* dst = Bs + r * 32 + c;
      copy16_async(src, dst);
      copy16_async(src + 8, dst + 8);
    }
    if (ks + 1 < KD / 32) {  // -> global_prefetch_b8
      __builtin_prefetch(A + (size_t)(m0 + r) * KD + (ks + 1) * 32, 0, 1);
      __builtin_prefetch(Wt + (size_t)(n0 + r) * KD + (ks + 1) * 32, 0, 1);
    }
    wait_async0();
    __syncthreads();
    v16bf af = frag_a(As + (w * 16) * 32, 32);
#pragma unroll
    for (int nt = 0; nt < 8; ++nt) {
      v16bf bfg = frag_a(Bs + (nt * 16) * 32, 32);
      acc[nt] = wmma_bf16(af, bfg, acc[nt]);
    }
    __syncthreads();
  }
}

// ---------------------------------------------------------------------------
// QKV GEMM: Xb[8192,768] @ Wt[2304,768]^T + bias
//   -> Q,K bf16 in [B,H,S,HD];  V bf16 in [B,H,HD,S] (transposed for attn)
// ---------------------------------------------------------------------------
__global__ __launch_bounds__(256)
void qkv_gemm_kernel(const bf16_t* __restrict__ Xb,
                     const bf16_t* __restrict__ Wt,   // [N3D, D] (transposed)
                     const float* __restrict__ bias,
                     bf16_t* __restrict__ q_out,
                     bf16_t* __restrict__ k_out,
                     bf16_t* __restrict__ v_out) {
  __shared__ bf16_t As[128 * 32];
  __shared__ bf16_t Bs[128 * 32];
  const int tiles_n = N3D / 128;  // 18
  const int m0 = (blockIdx.x / tiles_n) * 128;
  const int n0 = (blockIdx.x % tiles_n) * 128;
  const int t = threadIdx.x, w = t >> 5, l = t & 31;

  v8f acc[8];
#pragma unroll
  for (int i = 0; i < 8; ++i) acc[i] = v8f{};

  gemm_core<D_>(Xb, Wt, As, Bs, m0, n0, acc);

  // C-layout element (e, lane): row = m0+16w+e+8*(l>>4), col = n0+nt*16+(l&15)
  const int mbase = m0 + w * 16 + ((l >> 4) << 3);
  const int b  = mbase >> 11;       // row block never straddles batch (128|2048)
  const int s0 = mbase & 2047;
  const int nlane = n0 + (l & 15);
#pragma unroll
  for (int nt = 0; nt < 8; ++nt) {
    int n   = nlane + nt * 16;
    int sel = n / D_;               // uniform per tile chunk (768 % 128 == 0)
    int d   = n - sel * D_;
    int h   = d >> 6;
    int hd  = d & 63;
    float bv = bias[n];
    if (sel == 2) {
      // V transposed: [B,H,HD,S]; e runs along s -> contiguous stores
      bf16_t* p = v_out + ((size_t)(b * H_ + h) * HD_ + hd) * S_ + s0;
#pragma unroll
      for (int e = 0; e < 8; ++e) p[e] = (bf16_t)(acc[nt][e] + bv);
    } else {
      bf16_t* base = (sel == 0) ? q_out : k_out;   // [B,H,S,HD]
      bf16_t* p = base + ((size_t)(b * H_ + h) * S_ + s0) * HD_ + hd;
#pragma unroll
      for (int e = 0; e < 8; ++e) p[(size_t)e * HD_] = (bf16_t)(acc[nt][e] + bv);
    }
  }
}

// ---------------------------------------------------------------------------
// Fused attention: out = relu(QK^T/8)^2 @ V, per (b,h), 128 queries / block.
// No softmax -> pure accumulation over key tiles, no rescaling passes.
// ---------------------------------------------------------------------------
__global__ __launch_bounds__(256)
void attn_kernel(const bf16_t* __restrict__ Qg,   // [B,H,S,HD]
                 const bf16_t* __restrict__ Kg,   // [B,H,S,HD]
                 const bf16_t* __restrict__ Vg,   // [B,H,HD,S]  (transposed)
                 bf16_t* __restrict__ Og) {       // [B*S, 768] bf16
  __shared__ bf16_t Kt[32 * 64];        // [key][hd]  (== B^T layout for QK^T)
  __shared__ bf16_t Vt[64 * 32];        // [hd][key]  (== B^T layout for P*V)
  __shared__ bf16_t Ps[8][16 * 32];     // wave-private P bounce buffers

  const int t = threadIdx.x, w = t >> 5, l = t & 31;
  const int qt = blockIdx.x & 15;       // S/128 = 16 query tiles
  const int bh = blockIdx.x >> 4;       // 0..47
  const int b = bh / H_, h = bh % H_;

  const bf16_t* Qb = Qg + (size_t)bh * S_ * HD_;
  const bf16_t* Kb = Kg + (size_t)bh * S_ * HD_;
  const bf16_t* Vb = Vg + (size_t)bh * HD_ * S_;
  const int q0 = qt * 128 + w * 16;     // this wave's 16-query strip

  // Q fragments for hd 0..31 and 32..63 (held for the whole kernel)
  v16bf qf0 = frag_a(Qb + (size_t)q0 * HD_, HD_);
  v16bf qf1 = frag_a(Qb + (size_t)q0 * HD_ + 32, HD_);

  v8f o[4];
#pragma unroll
  for (int i = 0; i < 4; ++i) o[i] = v8f{};

  // per-thread tile-copy coordinates (8 bf16 = 16B chunks)
  const int vrow = t >> 2;              // hd row 0..63
  const int vcol = (t & 3) * 8;         // key col chunk

  for (int kt = 0; kt < S_ / 32; ++kt) {
    __syncthreads();  // protect Kt/Vt reuse
    const bf16_t* ksrc = Kb + (size_t)kt * 32 * HD_ + t * 8;   // contiguous tile
    bf16_t* kdst = Kt + t * 8;
    const bf16_t* vsrc = Vb + (size_t)vrow * S_ + kt * 32 + vcol;
    bf16_t* vdst = Vt + vrow * 32 + vcol;
    copy16_async(ksrc, kdst);
    copy16_async(vsrc, vdst);
    wait_async0();
    __syncthreads();

    // Scores for 16 q x 32 keys: two 16x16 n-tiles, K-dim = 64 (2 WMMAs each)
#pragma unroll
    for (int nt = 0; nt < 2; ++nt) {
      v8f sc = v8f{};
      sc = wmma_bf16(qf0, frag_a(Kt + (nt * 16) * 64, 64), sc);
      sc = wmma_bf16(qf1, frag_a(Kt + (nt * 16) * 64 + 32, 64), sc);
      // relu^2 of scaled scores, C-layout -> LDS (row-major P tile)
      const int col0 = nt * 16 + (l & 15);
      const int row0 = (l >> 4) << 3;
#pragma unroll
      for (int e = 0; e < 8; ++e) {
        float v = sc[e] * 0.125f;            // 1/sqrt(64)
        v = (v > 0.f) ? v * v : 0.f;
        Ps[w][(row0 + e) * 32 + col0] = (bf16_t)v;
      }
    }
    // intra-wave LDS RAW: wait for P stores before fragment reload
    asm volatile("s_wait_dscnt 0" ::: "memory");
    v16bf pf = frag_a(&Ps[w][0], 32);
#pragma unroll
    for (int nt = 0; nt < 4; ++nt) {
      o[nt] = wmma_bf16(pf, frag_a(Vt + (nt * 16) * 32, 32), o[nt]);
    }
  }

  // Write O (bf16) into [B*S, 768]: col = h*64 + nt*16 + (l&15)
  const int row0 = q0 + ((l >> 4) << 3);
  const int col0 = h * 64 + (l & 15);
#pragma unroll
  for (int nt = 0; nt < 4; ++nt) {
#pragma unroll
    for (int e = 0; e < 8; ++e) {
      int srow = row0 + e;
      Og[(size_t)(b * S_ + srow) * D_ + col0 + nt * 16] = (bf16_t)o[nt][e];
    }
  }
}

// ---------------------------------------------------------------------------
// Projection GEMM: AttnOut[8192,768] @ Wpt[768,768]^T + bias -> fp32 out
// ---------------------------------------------------------------------------
__global__ __launch_bounds__(256)
void proj_gemm_kernel(const bf16_t* __restrict__ Ab,
                      const bf16_t* __restrict__ Wt,   // [D, D] transposed
                      const float* __restrict__ bias,
                      float* __restrict__ out) {
  __shared__ bf16_t As[128 * 32];
  __shared__ bf16_t Bs[128 * 32];
  const int tiles_n = D_ / 128;  // 6
  const int m0 = (blockIdx.x / tiles_n) * 128;
  const int n0 = (blockIdx.x % tiles_n) * 128;
  const int t = threadIdx.x, w = t >> 5, l = t & 31;

  v8f acc[8];
#pragma unroll
  for (int i = 0; i < 8; ++i) acc[i] = v8f{};

  gemm_core<D_>(Ab, Wt, As, Bs, m0, n0, acc);

  const int mbase = m0 + w * 16 + ((l >> 4) << 3);
  const int nlane = n0 + (l & 15);
#pragma unroll
  for (int nt = 0; nt < 8; ++nt) {
    int n = nlane + nt * 16;
    float bv = bias[n];
#pragma unroll
    for (int e = 0; e < 8; ++e) {
      out[(size_t)(mbase + e) * D_ + n] = acc[nt][e] + bv;
    }
  }
}

// ---------------------------------------------------------------------------
// Host launcher
// ---------------------------------------------------------------------------
extern "C" void kernel_launch(void* const* d_in, const int* in_sizes, int n_in,
                              void* d_out, int out_size, void* d_ws, size_t ws_size,
                              hipStream_t stream) {
  const float* X     = (const float*)d_in[0];  // [B,S,D]
  const float* Wattn = (const float*)d_in[1];  // [D, 3D]
  const float* battn = (const float*)d_in[2];  // [3D]
  const float* Wproj = (const float*)d_in[3];  // [D, D]
  const float* bproj = (const float*)d_in[4];  // [D]
  float* out = (float*)d_out;

  // Carve workspace (256B-aligned chunks)
  char* ws = (char*)d_ws;
  size_t off = 0;
  auto carve = [&](size_t bytes) -> bf16_t* {
    bf16_t* p = (bf16_t*)(ws + off);
    off += (bytes + 255) & ~(size_t)255;
    return p;
  };
  bf16_t* Xb  = carve((size_t)BS * D_ * 2);     // X in bf16
  bf16_t* Wat = carve((size_t)D_ * N3D * 2);    // attn weight bf16, [N3D, D]
  bf16_t* Wpt = carve((size_t)D_ * D_ * 2);     // proj weight bf16, [D, D] (tr)
  bf16_t* Qw  = carve((size_t)BS * D_ * 2);     // [B,H,S,HD]
  bf16_t* Kw  = carve((size_t)BS * D_ * 2);     // [B,H,S,HD]
  bf16_t* Vw  = carve((size_t)BS * D_ * 2);     // [B,H,HD,S]
  bf16_t* Aw  = carve((size_t)BS * D_ * 2);     // attention output [B*S, D]

  // 1) fp32 -> bf16 conversions (weights transposed once)
  cvt_bf16_kernel<<<(BS * D_ / 4 + 255) / 256, 256, 0, stream>>>(X, Xb, BS * D_);
  cvt_bf16_tr_kernel<<<(D_ * N3D + 255) / 256, 256, 0, stream>>>(Wattn, Wat, D_, N3D);
  cvt_bf16_tr_kernel<<<(D_ * D_ + 255) / 256, 256, 0, stream>>>(Wproj, Wpt, D_, D_);

  // 2) QKV GEMM (64 x 18 tiles of 128x128)
  qkv_gemm_kernel<<<(BS / 128) * (N3D / 128), 256, 0, stream>>>(
      Xb, Wat, battn, Qw, Kw, Vw);

  // 3) Fused squared-ReLU attention (B*H * S/128 blocks)
  attn_kernel<<<B_ * H_ * (S_ / 128), 256, 0, stream>>>(Qw, Kw, Vw, Aw);

  // 4) Output projection (64 x 6 tiles of 128x128), fp32 out + bias
  proj_gemm_kernel<<<(BS / 128) * (D_ / 128), 256, 0, stream>>>(
      Aw, Wpt, bproj, out);
}
